// EarlyExitWeightedModel_38500086841781
// MI455X (gfx1250) — matrix operand order repaired and checked by reading, not compiled, not run on hardware
//
#include <hip/hip_runtime.h>
#include <hip/hip_bf16.h>
#include <math.h>

#define B_   16
#define T_   32
#define L_   128
#define D_   768
#define NH_  12
#define HD_  64
#define KSTEPS (D_ / 32)      // 24 k-steps of 32 per WMMA
#define MT  ((B_ * T_) / 16)  // 32 M tiles
#define NT  (D_ / 16)         // 48 N tiles
#define NEGV (-1e9f)

typedef __attribute__((ext_vector_type(16))) __bf16 bf16x16;
typedef __attribute__((ext_vector_type(8)))  float  f32x8;

static __device__ __forceinline__ unsigned short f2bf(float f) {
  unsigned int u = __float_as_uint(f);
  u += 0x7FFFu + ((u >> 16) & 1u);            // round-to-nearest-even
  return (unsigned short)(u >> 16);
}

// ---------------------------------------------------------------------------
// 1) Embedding gather + masked mean pool: H[b,t,:] (one block per (b,t))
// ---------------------------------------------------------------------------
__global__ void __launch_bounds__(256)
k_pool(const int* __restrict__ ids, const float* __restrict__ am,
       const float* __restrict__ embed, float* __restrict__ H) {
  __shared__ int   lid[L_];
  __shared__ float lm[L_];
  __shared__ float cnts;
  const int bt  = blockIdx.x;
  const int tid = threadIdx.x;
  const size_t base = (size_t)bt * L_;
  if (tid < L_) { lid[tid] = ids[base + tid]; lm[tid] = am[base + tid]; }
  __syncthreads();
  if (tid == 0) {
    float c = 0.f;
    for (int l = 0; l < L_; ++l) c += lm[l];
    cnts = fmaxf(c, 1.0f);                    // jnp.clip(.., 1.0)
  }
  __syncthreads();
  const float inv = 1.0f / cnts;
  for (int d = tid; d < D_; d += 256) {
    float acc = 0.f;
    for (int l = 0; l < L_; ++l) {
      if (l + 1 < L_)
        __builtin_prefetch(embed + (size_t)lid[l + 1] * D_ + d, 0, 0);
      acc = fmaf(embed[(size_t)lid[l] * D_ + d], lm[l], acc);
    }
    H[(size_t)bt * D_ + d] = acc * inv;
  }
}

// ---------------------------------------------------------------------------
// 2) Pack fp32 activations into per-lane WMMA A-fragment order (bf16)
//    layout: [mtile][kstep][lane(32)][elem(16)]   (ISA 05_wmma §7.12.2)
// ---------------------------------------------------------------------------
__global__ void __launch_bounds__(256)
k_pack_a(const float* __restrict__ src, unsigned short* __restrict__ dst) {
  const int idx = blockIdx.x * 256 + threadIdx.x;
  if (idx >= MT * KSTEPS * 512) return;
  const int e     = idx & 15;
  const int lane  = (idx >> 4) & 31;
  const int blk   = idx >> 9;
  const int ks    = blk % KSTEPS;
  const int mtile = blk / KSTEPS;
  const int h = lane >> 4;
  const int m = mtile * 16 + (lane & 15);
  const int j = e >> 1, p = e & 1;
  const int kk = ks * 32 +
      ((j < 4) ? (2 * j + p + 8 * h) : (16 + 2 * (j - 4) + p + 8 * h));
  dst[idx] = f2bf(src[(size_t)m * D_ + kk]);
}

// Pack fp32 weight W[K,N] into per-lane WMMA B-fragment order (bf16)
//    layout: [ntile][kstep][lane(32)][elem(16)], k = ks*32 + (e<8 ? 8*(lane/16)+e : 16+8*(lane/16)+e-8)
__global__ void __launch_bounds__(256)
k_pack_b(const float* __restrict__ W, unsigned short* __restrict__ dst) {
  const int idx = blockIdx.x * 256 + threadIdx.x;
  if (idx >= NT * KSTEPS * 512) return;
  const int e     = idx & 15;
  const int lane  = (idx >> 4) & 31;
  const int blk   = idx >> 9;
  const int ks    = blk % KSTEPS;
  const int ntile = blk / KSTEPS;
  const int n = ntile * 16 + (lane & 15);
  const int k = ks * 32 + ((e < 8) ? (8 * (lane >> 4) + e) : (16 + 8 * (lane >> 4) + (e - 8)));
  dst[idx] = f2bf(W[(size_t)k * D_ + n]);
}

// ---------------------------------------------------------------------------
// 3) WMMA GEMM: Out[mat] = A @ W[mat] + bias[mat].
//    4 waves / block, one 16x16 tile per wave, batched over gridDim.z mats.
//    zeroT0: zero entire output row where (m % T_) == 0 (after bias).
// ---------------------------------------------------------------------------
__global__ void __launch_bounds__(128)
k_gemm(const unsigned short* __restrict__ Apack,
       const unsigned short* __restrict__ Wpack, long wStride,
       const float* __restrict__ b0, const float* __restrict__ b1,
       const float* __restrict__ b2,
       float* __restrict__ Out, long oStride, int zeroT0) {
  const int lane  = threadIdx.x & 31;
  const int wv    = threadIdx.x >> 5;
  const int ntile = blockIdx.x * 4 + wv;
  const int mtile = blockIdx.y;
  const int mat   = blockIdx.z;

  const unsigned short* Ap = Apack + (size_t)mtile * KSTEPS * 512;
  const unsigned short* Bp = Wpack + (size_t)mat * (size_t)wStride +
                             (size_t)ntile * KSTEPS * 512;
  f32x8 acc = {};
#pragma unroll 4
  for (int ks = 0; ks < KSTEPS; ++ks) {
    bf16x16 a = *(const bf16x16*)(Ap + ks * 512 + lane * 16);
    bf16x16 b = *(const bf16x16*)(Bp + ks * 512 + lane * 16);
    acc = __builtin_amdgcn_wmma_f32_16x16x32_bf16(
        false, a, false, b, (short)0, acc, false, false);
  }
  const float* bias = (mat == 0) ? b0 : ((mat == 1) ? b1 : b2);
  const int n = ntile * 16 + (lane & 15);
  const int h = lane >> 4;
  const float bvv = bias[n];
  float* out = Out + (size_t)mat * (size_t)oStride;
  const int mbase = mtile * 16 + 8 * h;
#pragma unroll
  for (int r = 0; r < 8; ++r) {
    const int m = mbase + r;
    float val = acc[r] + bvv;
    if (zeroT0 && ((m & (T_ - 1)) == 0)) val = 0.f;   // t == 0 row
    out[(size_t)m * D_ + n] = val;
  }
}

// ---------------------------------------------------------------------------
// 4) Cross-turn causal attention, one block = one (b, head). T_=32 == wave32.
// ---------------------------------------------------------------------------
__global__ void __launch_bounds__(64)
k_attn(const float* __restrict__ qb, const float* __restrict__ kb,
       const float* __restrict__ vb, const float* __restrict__ turn_mask,
       float* __restrict__ ctx) {
  __shared__ float qsh[T_ * HD_], ksh[T_ * HD_], vsh[T_ * HD_];
  __shared__ float ash[T_ * T_];
  const int b  = blockIdx.x / NH_;
  const int hh = blockIdx.x % NH_;
  const int tid = threadIdx.x;
  for (int idx = tid; idx < T_ * HD_; idx += 64) {
    const int t = idx >> 6, d = idx & 63;
    const size_t g = ((size_t)(b * T_ + t)) * D_ + hh * HD_ + d;
    qsh[idx] = qb[g]; ksh[idx] = kb[g]; vsh[idx] = vb[g];
  }
  __syncthreads();
  if (tid < 32) {               // wave 0: lane j owns key j
    const int j = tid;
    const float tmj = turn_mask[b * T_ + j];
    for (int i = 0; i < T_; ++i) {
      float dot = 0.f;
      for (int d = 0; d < HD_; ++d)
        dot = fmaf(qsh[i * HD_ + d], ksh[j * HD_ + d], dot);
      float sc = ((j < i) && (tmj > 0.f)) ? dot * 0.125f : NEGV; // 1/sqrt(64)
      float mx = sc;
      for (int off = 16; off; off >>= 1)
        mx = fmaxf(mx, __shfl_xor(mx, off, 32));
      const float e = __expf(sc - mx);
      float se = e;
      for (int off = 16; off; off >>= 1) se += __shfl_xor(se, off, 32);
      ash[i * T_ + j] = e / se;
    }
  }
  __syncthreads();
  {                             // all 64 threads: thread d owns dim d
    const int d = tid;
    for (int i = 0; i < T_; ++i) {
      float s = 0.f;
      for (int j = 0; j < T_; ++j)
        s = fmaf(ash[i * T_ + j], vsh[j * HD_ + d], s);
      ctx[((size_t)(b * T_ + i)) * D_ + hh * HD_ + d] = s;
    }
  }
}

// ---------------------------------------------------------------------------
// 5) Evidence head + noisy-or recurrence. One block per dialogue b.
//    out = [qprob (B*T), p_agg (B*T), p_dialogue (B)]
// ---------------------------------------------------------------------------
__global__ void __launch_bounds__(256)
k_head(const float* __restrict__ H, const float* __restrict__ ctxO,
       const float* __restrict__ Wh, const float* __restrict__ bh,
       const float* __restrict__ turn_mask, float* __restrict__ out) {
  __shared__ float wsum[8];
  __shared__ float sarr[T_];
  const int b = blockIdx.x, tid = threadIdx.x;
  const int lane = tid & 31, wv = tid >> 5;
  for (int t = 0; t < T_; ++t) {
    float partial = 0.f;
    const size_t row = (size_t)(b * T_ + t) * D_;
    for (int idx = tid; idx < 2 * D_; idx += 256) {
      const float zv = (idx < D_) ? H[row + idx] : ctxO[row + idx - D_];
      partial = fmaf(zv, Wh[idx], partial);
    }
    for (int off = 16; off; off >>= 1) partial += __shfl_xor(partial, off, 32);
    if (lane == 0) wsum[wv] = partial;
    __syncthreads();
    if (tid == 0) {
      float tot = 0.f;
      for (int w = 0; w < 8; ++w) tot += wsum[w];
      sarr[t] = tot + bh[0];
    }
    __syncthreads();
  }
  if (tid == 0) {
    float prod = 1.0f;
    for (int t = 0; t < T_; ++t) {
      const float q =
          (1.0f / (1.0f + __expf(-sarr[t]))) * turn_mask[b * T_ + t];
      out[b * T_ + t] = q;                           // qprob
      prod *= (1.0f - q);
      out[B_ * T_ + b * T_ + t] = 1.0f - prod;       // p_agg
    }
    out[2 * B_ * T_ + b] = 1.0f - prod;              // p_dialogue
  }
}

// ---------------------------------------------------------------------------
extern "C" void kernel_launch(void* const* d_in, const int* in_sizes, int n_in,
                              void* d_out, int out_size, void* d_ws,
                              size_t ws_size, hipStream_t stream) {
  const int*   ids = (const int*)  d_in[0];
  const float* am  = (const float*)d_in[1];
  const float* tm  = (const float*)d_in[2];
  const float* emb = (const float*)d_in[3];
  const float* Wq  = (const float*)d_in[4];
  const float* bq  = (const float*)d_in[5];
  const float* Wk  = (const float*)d_in[6];
  const float* bk  = (const float*)d_in[7];
  const float* Wv  = (const float*)d_in[8];
  const float* bv  = (const float*)d_in[9];
  const float* Wo  = (const float*)d_in[10];
  const float* bo  = (const float*)d_in[11];
  const float* Wh  = (const float*)d_in[12];
  const float* bh  = (const float*)d_in[13];
  float* out = (float*)d_out;

  const size_t HN = (size_t)B_ * T_ * D_;  // 393216 activations
  const size_t WN = (size_t)D_ * D_;       // 589824 weight elements

  float* H    = (float*)d_ws;
  float* qb   = H    + HN;
  float* kb   = qb   + HN;
  float* vbuf = kb   + HN;
  float* ctx  = vbuf + HN;
  float* ctxO = ctx  + HN;
  unsigned short* Apack = (unsigned short*)(ctxO + HN);  // H  in A-frag order
  unsigned short* Cpack = Apack + HN;                    // ctx in A-frag order
  unsigned short* Wpack = Cpack + HN;                    // 4 * WN (q,k,v,o)

  // 1) pool
  k_pool<<<B_ * T_, 256, 0, stream>>>(ids, am, emb, H);
  // 2) pack operands
  k_pack_a<<<(int)(HN / 256), 256, 0, stream>>>(H, Apack);
  k_pack_b<<<(int)(WN / 256), 256, 0, stream>>>(Wq, Wpack + 0 * WN);
  k_pack_b<<<(int)(WN / 256), 256, 0, stream>>>(Wk, Wpack + 1 * WN);
  k_pack_b<<<(int)(WN / 256), 256, 0, stream>>>(Wv, Wpack + 2 * WN);
  k_pack_b<<<(int)(WN / 256), 256, 0, stream>>>(Wo, Wpack + 3 * WN);
  // 3) fused Q/K/V projections (batched over z)
  k_gemm<<<dim3(NT / 4, MT, 3), 128, 0, stream>>>(
      Apack, Wpack, (long)WN, bq, bk, bv, qb, (long)HN, 0);
  // 4) attention
  k_attn<<<B_ * NH_, 64, 0, stream>>>(qb, kb, vbuf, tm, ctx);
  // 5) output projection (+bias, zero t==0 rows)
  k_pack_a<<<(int)(HN / 256), 256, 0, stream>>>(ctx, Cpack);
  k_gemm<<<dim3(NT / 4, MT, 1), 128, 0, stream>>>(
      Cpack, Wpack + 3 * WN, (long)WN, bo, bo, bo, ctxO, (long)HN, 1);
  // 6) head + noisy-or
  k_head<<<B_, 256, 0, stream>>>(H, ctxO, Wh, bh, tm, out);
}
